// TVA_59090160059166
// MI455X (gfx1250) — matrix operand order repaired and checked by reading, not compiled, not run on hardware
//
#include <hip/hip_runtime.h>
#include <hip/hip_bf16.h>
#include <math.h>

// ---------------------------------------------------------------------------
// Problem constants (from reference): B=128, L=256, D=512, H=8, NL=2, TOPK=110
// ---------------------------------------------------------------------------
#define BB   128
#define LL   256
#define DD   512
#define BL   (BB * LL)          // 32768
#define TOPK 110

typedef __attribute__((ext_vector_type(16))) _Float16 v16h;
typedef __attribute__((ext_vector_type(8)))  _Float16 v8h;
typedef __attribute__((ext_vector_type(8)))  float    v8f;

// ---------------------------------------------------------------------------
// Fused activation
// ---------------------------------------------------------------------------
__device__ __forceinline__ float gelu_tanh(float x) {
    const float c = 0.7978845608028654f; // sqrt(2/pi)
    float u = c * (x + 0.044715f * x * x * x);
    return 0.5f * x * (1.0f + tanhf(u));
}

template <int ACT>
__device__ __forceinline__ float act_apply(float x) {
    if constexpr (ACT == 1) return fmaxf(x, 0.0f);
    if constexpr (ACT == 2) return gelu_tanh(x);
    return x;
}

// ---------------------------------------------------------------------------
// WMMA GEMM:  Cout = act( A[MxK] * B + bias + Cin )
//   !TRANSB: B is row-major K x N      (weights)
//    TRANSB: B is row-major N x K      (C = A * B^T, used for S = Q K^T)
//
// Block tile 128x128, 8 waves arranged 4(M) x 2(N); each wave owns a 32x64
// sliver = 2 A-frags x 4 B-frags = 8 v_wmma_f32_16x16x32_f16 per K-step.
// f32 -> f16 conversion happens while staging tiles into LDS; accumulation
// stays f32.  LDS rows are padded to 80B so every fragment read is a 16B-
// aligned ds_load_b128.
// ---------------------------------------------------------------------------
#define TILE_M 128
#define TILE_N 128
#define TILE_K 32
#define LDSW   40   // halfs per LDS row: 32 + 8 pad -> 80B (16B aligned rows)

template <bool TRANSB, int ACT>
__global__ __launch_bounds__(256)
void gemm_f16w(const float* __restrict__ A, const float* __restrict__ Bm,
               const float* __restrict__ bias, const float* __restrict__ Cin,
               float* __restrict__ Cout,
               int M, int N, int K, int lda, int ldb, int ldc,
               long batchA, long batchB, long batchC)
{
    __shared__ __align__(16) _Float16 As[TILE_M * LDSW];
    __shared__ __align__(16) _Float16 Bs[TILE_N * LDSW];

    const int bz = blockIdx.z;
    A    += (long)bz * batchA;
    Bm   += (long)bz * batchB;
    if (Cin) Cin += (long)bz * batchC;
    Cout += (long)bz * batchC;

    const int tid  = threadIdx.x;
    const int lane = tid & 31;
    const int wave = tid >> 5;
    const int wm   = wave & 3;   // 0..3 : 32-row sliver
    const int wn   = wave >> 2;  // 0..1 : 64-col sliver

    const int m0 = blockIdx.y * TILE_M;
    const int n0 = blockIdx.x * TILE_N;

    v8f acc[2][4] = {};

    // staging indices: each thread converts 16 contiguous f32 -> f16
    const int arow = tid >> 1;         // 0..127
    const int acol = (tid & 1) * 16;   // 0 or 16
    const int bkr  = tid >> 3;         // 0..31  (K row, !TRANSB)
    const int bn16 = (tid & 7) * 16;   // N groups of 16

    // fragment bases (ISA 7.12.2 layouts, wave32)
    const int fab = (lane < 16) ? 0 : 8;    // A: K 0-7/16-23 vs 8-15/24-31
    const int fbk = (lane < 16) ? 0 : 8;    // B: K 0-7/16-23 vs 8-15/24-31
    const int l15 = lane & 15;

    for (int k0 = 0; k0 < K; k0 += TILE_K) {
        // ---- stage A tile (128 x 32) as f16 ----
        {
            const float* g = A + (long)(m0 + arow) * lda + (k0 + acol);
            _Float16* s = &As[arow * LDSW + acol];
            #pragma unroll
            for (int j = 0; j < 16; ++j) s[j] = (_Float16)g[j];
            if (k0 + TILE_K < K)
                __builtin_prefetch(g + TILE_K, 0, 0);   // global_prefetch_b8
        }
        // ---- stage B tile into Bs[n][k] ----
        if constexpr (!TRANSB) {
            const float* g = Bm + (long)(k0 + bkr) * ldb + (n0 + bn16);
            #pragma unroll
            for (int j = 0; j < 16; ++j)
                Bs[(bn16 + j) * LDSW + bkr] = (_Float16)g[j];
            if (k0 + TILE_K < K)
                __builtin_prefetch(g + (long)TILE_K * ldb, 0, 0);
        } else {
            const float* g = Bm + (long)(n0 + arow) * ldb + (k0 + acol);
            _Float16* s = &Bs[arow * LDSW + acol];
            #pragma unroll
            for (int j = 0; j < 16; ++j) s[j] = (_Float16)g[j];
            if (k0 + TILE_K < K)
                __builtin_prefetch(g + TILE_K, 0, 0);
        }
        __syncthreads();

        // ---- load fragments (two 16B-aligned ds_load_b128 each) ----
        union F { v16h v; v8h h[2]; };
        F a[2], bf[4];
        #pragma unroll
        for (int i = 0; i < 2; ++i) {
            const _Float16* ap = &As[(wm * 32 + i * 16 + l15) * LDSW + fab];
            a[i].h[0] = *(const v8h*)(ap);
            a[i].h[1] = *(const v8h*)(ap + 16);
        }
        #pragma unroll
        for (int j = 0; j < 4; ++j) {
            const _Float16* bp = &Bs[(wn * 64 + j * 16 + l15) * LDSW + fbk];
            bf[j].h[0] = *(const v8h*)(bp);
            bf[j].h[1] = *(const v8h*)(bp + 16);
        }
        #pragma unroll
        for (int i = 0; i < 2; ++i)
            #pragma unroll
            for (int j = 0; j < 4; ++j)
                acc[i][j] = __builtin_amdgcn_wmma_f32_16x16x32_f16(
                    false, a[i].v, false, bf[j].v, (short)0, acc[i][j],
                    false, false);
        __syncthreads();
    }

    // ---- fused epilogue: bias + residual + activation ----
    const int rbase = m0 + wm * 32 + ((lane < 16) ? 0 : 8);
    const int cbase = n0 + wn * 64 + l15;
    #pragma unroll
    for (int i = 0; i < 2; ++i) {
        #pragma unroll
        for (int j = 0; j < 4; ++j) {
            const int col = cbase + j * 16;
            const float bia = bias ? bias[col] : 0.0f;
            #pragma unroll
            for (int r = 0; r < 8; ++r) {
                const long row = rbase + i * 16 + r;
                const long idx = row * (long)ldc + col;
                float v = acc[i][j][r] + bia;
                if (Cin) v += Cin[idx];
                Cout[idx] = act_apply<ACT>(v);
            }
        }
    }
}

// ---------------------------------------------------------------------------
// Embedding + time-interval affine:  xe = token[item]+pos ; ti0 = t*wi + bi
// ---------------------------------------------------------------------------
__global__ __launch_bounds__(256)
void embed_kernel(const int* __restrict__ item, const float* __restrict__ tim,
                  const float* __restrict__ token, const float* __restrict__ pos,
                  const float* __restrict__ tiw, const float* __restrict__ tib,
                  float* __restrict__ xe, float* __restrict__ ti0)
{
    const long i = (long)blockIdx.x * 256 + threadIdx.x;  // over B*L*D
    const int  d  = (int)(i & (DD - 1));
    const long bl = i >> 9;
    const int  l  = (int)(bl & (LL - 1));
    const int  it = item[bl];
    xe[i]  = token[(long)it * DD + d] + pos[l * DD + d];
    ti0[i] = tim[bl] * tiw[d] + tib[d];
}

// x[b,l,:] += y[b,:]
__global__ __launch_bounds__(256)
void bcast_add_kernel(float* __restrict__ x, const float* __restrict__ y)
{
    const long i = (long)blockIdx.x * 256 + threadIdx.x;
    x[i] += y[((i >> 17) << 9) + (i & (DD - 1))];
}

// ---------------------------------------------------------------------------
// LayerNorm over D=512, one block (256 threads) per row, matches reference
// (mean, then mean of squared deviation, rsqrt(var+1e-6))
// ---------------------------------------------------------------------------
__global__ __launch_bounds__(256)
void ln_kernel(const float* __restrict__ x, const float* __restrict__ g,
               const float* __restrict__ b, float* __restrict__ out)
{
    __shared__ float red[256];
    const long row = blockIdx.x;
    const float* xr = x + row * DD;
    const int t = threadIdx.x;
    const float v0 = xr[t], v1 = xr[t + 256];
    red[t] = v0 + v1; __syncthreads();
    for (int off = 128; off > 0; off >>= 1) {
        if (t < off) red[t] += red[t + off];
        __syncthreads();
    }
    const float mean = red[0] * (1.0f / DD);
    __syncthreads();
    const float d0 = v0 - mean, d1 = v1 - mean;
    red[t] = d0 * d0 + d1 * d1; __syncthreads();
    for (int off = 128; off > 0; off >>= 1) {
        if (t < off) red[t] += red[t + off];
        __syncthreads();
    }
    const float rs = rsqrtf(red[0] * (1.0f / DD) + 1e-6f);
    out[row * DD + t]       = g[t] * d0 * rs + b[t];
    out[row * DD + t + 256] = g[t + 256] * d1 * rs + b[t + 256];
}

// ---------------------------------------------------------------------------
// mean_value[b,tau] = (1/D) * sum_t S[b, t, (t - tau) mod L]
// (circular cross-correlation averaged over all D channels == diag sums of QK^T)
// ---------------------------------------------------------------------------
__global__ __launch_bounds__(256)
void diag_kernel(const float* __restrict__ S, float* __restrict__ mv)
{
    const int b = blockIdx.x, tau = threadIdx.x;
    const float* Sb = S + (long)b * LL * LL;
    float s = 0.0f;
    for (int t = 0; t < LL; ++t)
        s += Sb[t * LL + ((t - tau) & (LL - 1))];
    mv[b * LL + tau] = s * (1.0f / DD);
}

// top-K(110) of mean over batch; stable (smallest index wins ties) like lax.top_k
__global__ __launch_bounds__(256)
void topk_kernel(const float* __restrict__ mv, int* __restrict__ idx_out)
{
    __shared__ float gv[256];
    __shared__ float rv[256];
    __shared__ int   ri[256];
    const int t = threadIdx.x;
    float s = 0.0f;
    for (int b = 0; b < BB; ++b) s += mv[b * LL + t];
    gv[t] = s;
    __syncthreads();
    for (int kk = 0; kk < TOPK; ++kk) {
        rv[t] = gv[t]; ri[t] = t; __syncthreads();
        for (int off = 128; off > 0; off >>= 1) {
            if (t < off) {
                if (rv[t + off] > rv[t] ||
                    (rv[t + off] == rv[t] && ri[t + off] < ri[t])) {
                    rv[t] = rv[t + off]; ri[t] = ri[t + off];
                }
            }
            __syncthreads();
        }
        if (t == 0) { idx_out[kk] = ri[0]; gv[ri[0]] = -3.4e38f; }
        __syncthreads();
    }
}

// per-batch softmax over the 110 gathered lag weights
__global__ __launch_bounds__(128)
void softw_kernel(const float* __restrict__ mv, const int* __restrict__ idx,
                  float* __restrict__ sw)
{
    __shared__ float red[128];
    const int b = blockIdx.x, t = threadIdx.x;
    const float v = (t < TOPK) ? mv[b * LL + idx[t]] : -3.4e38f;
    red[t] = v; __syncthreads();
    for (int off = 64; off > 0; off >>= 1) {
        if (t < off) red[t] = fmaxf(red[t], red[t + off]);
        __syncthreads();
    }
    const float mx = red[0]; __syncthreads();
    const float e = (t < TOPK) ? __expf(v - mx) : 0.0f;
    red[t] = e; __syncthreads();
    for (int off = 64; off > 0; off >>= 1) {
        if (t < off) red[t] += red[t + off];
        __syncthreads();
    }
    sw[b * 128 + t] = (t < TOPK) ? (e / red[0]) : 0.0f;
}

// Build dense delay matrix: M_b[t,s] = lagw_b[(s - t) mod L]  (overwrites S)
__global__ __launch_bounds__(256)
void buildM_kernel(const float* __restrict__ sw, const int* __restrict__ idx,
                   float* __restrict__ Mout)
{
    __shared__ float lagw[LL];
    const int b = blockIdx.x, t = threadIdx.x;
    lagw[t] = 0.0f; __syncthreads();
    if (t < TOPK) lagw[idx[t]] = sw[b * 128 + t];
    __syncthreads();
    float* Mb = Mout + (long)b * LL * LL;
    for (int tt = 0; tt < LL; ++tt)
        Mb[tt * LL + t] = lagw[(t - tt) & (LL - 1)];
}

// ---------------------------------------------------------------------------
// Host-side GEMM launcher
// ---------------------------------------------------------------------------
template <bool TB, int ACT>
static void launch_gemm(hipStream_t stream, const float* A, const float* Bm,
                        const float* bias, const float* Cin, float* Cout,
                        int M, int N, int K, int lda, int ldb, int ldc,
                        long bA, long bB, long bC, int batch)
{
    dim3 grid((unsigned)(N / TILE_N), (unsigned)(M / TILE_M), (unsigned)batch);
    gemm_f16w<TB, ACT><<<grid, dim3(256), 0, stream>>>(
        A, Bm, bias, Cin, Cout, M, N, K, lda, ldb, ldc, bA, bB, bC);
}

extern "C" void kernel_launch(void* const* d_in, const int* in_sizes, int n_in,
                              void* d_out, int out_size, void* d_ws, size_t ws_size,
                              hipStream_t stream)
{
    (void)in_sizes; (void)n_in; (void)out_size; (void)ws_size;

    // -------- inputs (setup_inputs() dict order, params flattened in order) --
    const int*   item  = (const int*)  d_in[0];
    const float* tim   = (const float*)d_in[1];
    const float* ulf   = (const float*)d_in[2];   // [B,1024]: mu | sigma
    const float* token = (const float*)d_in[3];
    const float* pos   = (const float*)d_in[4];
    const float* lat_w = (const float*)d_in[5];
    const float* lat_b = (const float*)d_in[6];
    const float* lat2w = (const float*)d_in[7];
    const float* lat2b = (const float*)d_in[8];
    const float* ti_w  = (const float*)d_in[9];
    const float* ti_b  = (const float*)d_in[10];
    const float* pw1w  = (const float*)d_in[11];
    const float* pw1b  = (const float*)d_in[12];
    const float* pw2w  = (const float*)d_in[13];
    const float* pw2b  = (const float*)d_in[14];
    const float* outw  = (const float*)d_in[15];  // [2048,512]
    const float* outb  = (const float*)d_in[16];
    const int LAYER0 = 17;                        // 16 tensors per layer

    // -------- workspace layout (floats) --------
    float* ws  = (float*)d_ws;
    const long NX = (long)BL * DD;                // 16,777,216
    float* X   = ws;                              // persistent residual stream
    float* Hh  = ws + NX;                         // LN output
    float* BIG = ws + 2 * NX;                     // 75,497,472-float region
    float* Q   = BIG;
    float* Kb  = BIG + NX;
    float* Vb  = BIG + 2 * NX;
    float* Sb  = BIG + 3 * NX;                    // [B,256,256] = 8,388,608
    float* AG  = Sb + (long)BB * LL * LL;
    float* T1  = BIG;                             // FFN hidden, aliases Q..AG
    float* XE  = Q;                               // pre-layer aliases
    float* TI  = Kb;
    float* TI0 = Vb;
    float* TM1 = AG;
    float* SMALL = BIG + (3 * NX + (long)BB * LL * LL + NX);
    float* Y    = SMALL;                          // [128,512] latent sum
    float* TMPY = SMALL + 65536;                  // [128,512] latent temp
    float* MV   = SMALL + 131072;                 // [128,256]
    float* SW   = SMALL + 163840;                 // [128,128]
    int*   IDX  = (int*)(SMALL + 180224);         // [110]

    const int nblk = (int)(NX / 256);

    // -------- input path --------
    embed_kernel<<<nblk, 256, 0, stream>>>(item, tim, token, pos, ti_w, ti_b, XE, TI0);

    // ti = ti0 + relu(ti0@pw1+b)@pw2+b
    launch_gemm<false, 1>(stream, TI0, pw1w, pw1b, nullptr, TM1,
                          BL, DD, DD, DD, DD, DD, 0, 0, 0, 1);
    launch_gemm<false, 0>(stream, TM1, pw2w, pw2b, TI0, TI,
                          BL, DD, DD, DD, DD, DD, 0, 0, 0, 1);

    // y = mu@lat_w + lat_b + sigma@lat2_w + lat2_b   (mu/sigma strided in ulf)
    launch_gemm<false, 0>(stream, ulf, lat_w, lat_b, nullptr, Y,
                          BB, DD, DD, 1024, DD, DD, 0, 0, 0, 1);
    launch_gemm<false, 0>(stream, ulf + 512, lat2w, lat2b, Y, Y,
                          BB, DD, DD, 1024, DD, DD, 0, 0, 0, 1);

    // x = [xe|ti|lat|lat2] @ out_w + out_b  as 2 GEMMs + broadcast of y@slices
    launch_gemm<false, 0>(stream, XE, outw, outb, nullptr, X,
                          BL, DD, DD, DD, DD, DD, 0, 0, 0, 1);
    launch_gemm<false, 0>(stream, TI, outw + (long)DD * DD, nullptr, X, X,
                          BL, DD, DD, DD, DD, DD, 0, 0, 0, 1);
    // fold y through out_w slices 2,3 (small 128x512 GEMMs), then broadcast
    launch_gemm<false, 0>(stream, Y, outw + 2L * DD * DD, nullptr, nullptr, TMPY,
                          BB, DD, DD, DD, DD, DD, 0, 0, 0, 1);
    launch_gemm<false, 0>(stream, Y, outw + 3L * DD * DD, nullptr, TMPY, Y,
                          BB, DD, DD, DD, DD, DD, 0, 0, 0, 1);
    bcast_add_kernel<<<nblk, 256, 0, stream>>>(X, Y);

    // -------- transformer layers --------
    for (int layer = 0; layer < 2; ++layer) {
        const float* ln1g = (const float*)d_in[LAYER0 + 16 * layer + 0];
        const float* ln1b = (const float*)d_in[LAYER0 + 16 * layer + 1];
        const float* wq   = (const float*)d_in[LAYER0 + 16 * layer + 2];
        const float* bq   = (const float*)d_in[LAYER0 + 16 * layer + 3];
        const float* wk   = (const float*)d_in[LAYER0 + 16 * layer + 4];
        const float* bk   = (const float*)d_in[LAYER0 + 16 * layer + 5];
        const float* wv   = (const float*)d_in[LAYER0 + 16 * layer + 6];
        const float* bv   = (const float*)d_in[LAYER0 + 16 * layer + 7];
        const float* wo   = (const float*)d_in[LAYER0 + 16 * layer + 8];
        const float* bo   = (const float*)d_in[LAYER0 + 16 * layer + 9];
        const float* ln2g = (const float*)d_in[LAYER0 + 16 * layer + 10];
        const float* ln2b = (const float*)d_in[LAYER0 + 16 * layer + 11];
        const float* w1   = (const float*)d_in[LAYER0 + 16 * layer + 12];
        const float* b1   = (const float*)d_in[LAYER0 + 16 * layer + 13];
        const float* w2   = (const float*)d_in[LAYER0 + 16 * layer + 14];
        const float* b2   = (const float*)d_in[LAYER0 + 16 * layer + 15];

        ln_kernel<<<BL, 256, 0, stream>>>(X, ln1g, ln1b, Hh);
        launch_gemm<false, 0>(stream, Hh, wq, bq, nullptr, Q,  BL, DD, DD, DD, DD, DD, 0, 0, 0, 1);
        launch_gemm<false, 0>(stream, Hh, wk, bk, nullptr, Kb, BL, DD, DD, DD, DD, DD, 0, 0, 0, 1);
        launch_gemm<false, 0>(stream, Hh, wv, bv, nullptr, Vb, BL, DD, DD, DD, DD, DD, 0, 0, 0, 1);

        // S[b] = Q[b] @ K[b]^T   (batched, TRANSB)
        launch_gemm<true, 0>(stream, Q, Kb, nullptr, nullptr, Sb,
                             LL, LL, DD, DD, DD, LL,
                             (long)LL * DD, (long)LL * DD, (long)LL * LL, BB);
        diag_kernel <<<BB, 256, 0, stream>>>(Sb, MV);
        topk_kernel <<<1, 256, 0, stream>>>(MV, IDX);
        softw_kernel<<<BB, 128, 0, stream>>>(MV, IDX, SW);
        buildM_kernel<<<BB, 256, 0, stream>>>(SW, IDX, Sb);   // S := delay matrix

        // agg[b] = M[b] @ V[b]   (batched)
        launch_gemm<false, 0>(stream, Sb, Vb, nullptr, nullptr, AG,
                              LL, DD, LL, LL, DD, DD,
                              (long)LL * LL, (long)LL * DD, (long)LL * DD, BB);

        // x = x + agg @ wo + bo
        launch_gemm<false, 0>(stream, AG, wo, bo, X, X, BL, DD, DD, DD, DD, DD, 0, 0, 0, 1);

        // FFN: x = x + gelu(LN(x)@w1+b1)@w2+b2
        ln_kernel<<<BL, 256, 0, stream>>>(X, ln2g, ln2b, Hh);
        launch_gemm<false, 2>(stream, Hh, w1, b1, nullptr, T1,
                              BL, 4 * DD, DD, DD, 4 * DD, 4 * DD, 0, 0, 0, 1);
        float* dst = (layer == 1) ? (float*)d_out : X;
        launch_gemm<false, 0>(stream, T1, w2, b2, X, dst,
                              BL, DD, 4 * DD, 4 * DD, DD, DD, 0, 0, 0, 1);
    }
}
